// Model_37976100831499
// MI455X (gfx1250) — matrix operand order repaired and checked by reading, not compiled, not run on hardware
//
#include <hip/hip_runtime.h>

#define B_  2
#define S_  2048
#define E_  512
#define L_  32
#define V_  32000

typedef __bf16 v16bf __attribute__((ext_vector_type(16)));
typedef __bf16 v8bf  __attribute__((ext_vector_type(8)));
typedef float  v8f   __attribute__((ext_vector_type(8)));

union ABuild { v16bf v; v8bf h[2]; };

// ---------------------------------------------------------------------------
// Kernel 1: latent projections. 256 thr/block, 8 rows/block (32 latent dims).
// q is pre-scaled by 1/sqrt(32) so attention scores need no extra scaling.
// ---------------------------------------------------------------------------
__global__ __launch_bounds__(256) void k_proj(
    const float* __restrict__ Q, const float* __restrict__ K,
    const float* __restrict__ Vv, const float* __restrict__ X,
    const float* __restrict__ Wd, const float* __restrict__ bd,
    __bf16* __restrict__ qb, __bf16* __restrict__ kb,
    __bf16* __restrict__ vT, float* __restrict__ xl)
{
    const int r   = threadIdx.x >> 5;          // 0..7 row within block
    const int d   = threadIdx.x & 31;          // latent dim
    const int row = blockIdx.x * 8 + r;        // 0 .. B*S-1
    const float* q = Q  + (size_t)row * E_;
    const float* k = K  + (size_t)row * E_;
    const float* v = Vv + (size_t)row * E_;
    const float* x = X  + (size_t)row * E_;
    float sq = 0.f, sk = 0.f, sv = 0.f, sx = 0.f;
    #pragma unroll 4
    for (int e = 0; e < E_; ++e) {
        const float w = Wd[e * L_ + d];
        sq += q[e] * w; sk += k[e] * w; sv += v[e] * w; sx += x[e] * w;
    }
    const float bdv = bd[d];
    sq += bdv; sk += bdv; sv += bdv; sx += bdv;
    const int b = row / S_, s = row % S_;
    qb[(size_t)row * L_ + d] = (__bf16)(sq * 0.17677669529663687f); // 1/sqrt(32)
    kb[(size_t)row * L_ + d] = (__bf16)sk;
    vT[((size_t)b * L_ + d) * S_ + s] = (__bf16)sv;                 // transposed
    xl[(size_t)row * L_ + d] = sx;
}

// ---------------------------------------------------------------------------
// Kernel 2: W_out [512,32000] f32 -> WoT [32000,512] bf16.
// LDS tile staged with GLOBAL_LOAD_ASYNC_TO_LDS_B128 (ASYNCcnt path):
// 256 lanes x 16B = one 32x32 f32 tile. Row pad = 36 floats (144B) keeps every
// 16B chunk 16B-aligned in LDS (B128 alignment rule).
// ---------------------------------------------------------------------------
__global__ __launch_bounds__(256) void k_woutT(
    const float* __restrict__ Wo, __bf16* __restrict__ WoT)
{
    __shared__ float t[32][36];
    const int v0 = blockIdx.x * 32, k0 = blockIdx.y * 32;
    const int tx = threadIdx.x, ty = threadIdx.y;        // (32,8)
    const int tid = ty * 32 + tx;                        // 0..255

    // async copy: lane handles row (tid/8), 4 floats starting at col (tid%8)*4
    {
        const int rr = tid >> 3;
        const int cc = (tid & 7) * 4;
        const float* gsrc = Wo + (size_t)(k0 + rr) * V_ + v0 + cc;
        const uint32_t lds_off = (uint32_t)(uintptr_t)(&t[rr][cc]); // addr[31:0] = LDS offset
        const uint64_t gaddr   = (uint64_t)(uintptr_t)gsrc;
        asm volatile("global_load_async_to_lds_b128 %0, %1, off"
                     :: "v"(lds_off), "v"(gaddr) : "memory");
        asm volatile("s_wait_asynccnt 0x0" ::: "memory");
    }
    __syncthreads();

    #pragma unroll
    for (int i = 0; i < 4; ++i)
        WoT[(size_t)(v0 + ty + i * 8) * E_ + k0 + tx] = (__bf16)t[tx][ty + i * 8];
}

// ---------------------------------------------------------------------------
// Kernel 3: causal flash attention on latent-32, one wave per 16-query tile.
// S^T tiles via WMMA(A=K rows, B=Q^T); online softmax kept per-lane (lane=query),
// P re-used directly as next WMMA A-operand (in-lane exp + bf16 pack).
// ---------------------------------------------------------------------------
__global__ __launch_bounds__(32) void k_attn(
    const __bf16* __restrict__ qb, const __bf16* __restrict__ kbuf,
    const __bf16* __restrict__ vT, float* __restrict__ ctx)
{
    const int lane = threadIdx.x;
    const int half = lane >> 4;          // 0/1
    const int ln   = lane & 15;
    const int qt   = blockIdx.x;         // 0..127
    const int b    = blockIdx.y;         // 0..1
    const int qbase = qt * 16;
    const int qglob = qbase + ln;        // query tracked by this lane

    // B operand for S^T: lane ln holds Q row (qbase+ln), K-halves by `half`
    const v16bf Bq = *(const v16bf*)(qb + ((size_t)b * S_ + qbase + ln) * L_ + half * 16);

    v8f acc0 = {}, acc1 = {};
    float m = -1e30f, l = 0.f;

    for (int kb0 = 0; kb0 < qbase + 16; kb0 += 32) {
        // A operands: 2 tiles of 16 key rows each (16x32 bf16 A layout)
        const __bf16* kr0 = kbuf + ((size_t)b * S_ + kb0 + ln) * L_;
        const __bf16* kr1 = kbuf + ((size_t)b * S_ + kb0 + 16 + ln) * L_;
        ABuild A0, A1;
        A0.h[0] = *(const v8bf*)(kr0 + half * 16);
        A0.h[1] = *(const v8bf*)(kr0 + half * 16 + 8);
        A1.h[0] = *(const v8bf*)(kr1 + half * 16);
        A1.h[1] = *(const v8bf*)(kr1 + half * 16 + 8);
        const v8f z = {};
        v8f st0 = __builtin_amdgcn_wmma_f32_16x16x32_bf16(false, A0.v, false, Bq, (short)0, z, false, false);
        v8f st1 = __builtin_amdgcn_wmma_f32_16x16x32_bf16(false, A1.v, false, Bq, (short)0, z, false, false);

        // causal mask + running max (keys live in-lane: key = kb0 + t*16 + r + 8*half)
        float s0[8], s1[8], mloc = -1e30f;
        #pragma unroll
        for (int r = 0; r < 8; ++r) {
            const int key0 = kb0 + r + 8 * half;
            float a = st0[r]; if (key0 > qglob)      a = -1e30f;
            float c = st1[r]; if (key0 + 16 > qglob) c = -1e30f;
            s0[r] = a; s1[r] = c;
            mloc = fmaxf(mloc, fmaxf(a, c));
        }
        mloc = fmaxf(mloc, __shfl_xor(mloc, 16, 32));
        const float mnew  = fmaxf(m, mloc);
        const float alpha = __expf(m - mnew);

        // exp + pack P into WMMA A layout (pure in-lane)
        float lsum = 0.f;
        v16bf P;
        #pragma unroll
        for (int r = 0; r < 8; ++r) {
            const float p0 = __expf(s0[r] - mnew);
            const float p1 = __expf(s1[r] - mnew);
            lsum += p0 + p1;
            P[r]     = (__bf16)p0;
            P[r + 8] = (__bf16)p1;
        }
        lsum += __shfl_xor(lsum, 16, 32);
        l = l * alpha + lsum;
        m = mnew;

        // rescale accumulators (per-row alpha broadcast: row query = r + 8*half)
        #pragma unroll
        for (int r = 0; r < 8; ++r) {
            const float ar = __shfl(alpha, r + 8 * half, 32);
            acc0[r] *= ar;
            acc1[r] *= ar;
        }

        // B operands from V^T: lane ln -> latent dim (chunk*16 + ln), keys in table order (matching P)
        const __bf16* vr0 = vT + ((size_t)b * L_ + ln)      * S_ + kb0;
        const __bf16* vr1 = vT + ((size_t)b * L_ + 16 + ln) * S_ + kb0;
        ABuild Vu0, Vu1;
        Vu0.h[0] = *(const v8bf*)(vr0 + half * 8);
        Vu0.h[1] = *(const v8bf*)(vr0 + half * 8 + 16);
        Vu1.h[0] = *(const v8bf*)(vr1 + half * 8);
        Vu1.h[1] = *(const v8bf*)(vr1 + half * 8 + 16);
        const v16bf Vb0 = Vu0.v;
        const v16bf Vb1 = Vu1.v;
        acc0 = __builtin_amdgcn_wmma_f32_16x16x32_bf16(false, P, false, Vb0, (short)0, acc0, false, false);
        acc1 = __builtin_amdgcn_wmma_f32_16x16x32_bf16(false, P, false, Vb1, (short)0, acc1, false, false);
    }

    const float linv = 1.0f / l;
    #pragma unroll
    for (int r = 0; r < 8; ++r) {
        const float li = __shfl(linv, r + 8 * half, 32);
        float* out = ctx + ((size_t)b * S_ + qbase + r + 8 * half) * L_;
        out[ln]      = acc0[r] * li;
        out[16 + ln] = acc1[r] * li;
    }
}

// ---------------------------------------------------------------------------
// Kernel 4: up-projection + tiled-context residual + LayerNorm -> bf16 final.
// One block (512 thr) per token row.
// ---------------------------------------------------------------------------
__global__ __launch_bounds__(512) void k_upln(
    const float* __restrict__ ctx, const float* __restrict__ xl,
    const float* __restrict__ Wup, const float* __restrict__ bup,
    const float* __restrict__ lnw, const float* __restrict__ lnb,
    __bf16* __restrict__ fb)
{
    const int row = blockIdx.x;
    const int e   = threadIdx.x;
    __shared__ float xs[L_];
    __shared__ float red[16];
    if (e < L_) xs[e] = xl[(size_t)row * L_ + e];
    __syncthreads();

    float y = bup[e];
    #pragma unroll
    for (int d = 0; d < L_; ++d) y += xs[d] * Wup[d * E_ + e];
    y += ctx[(size_t)row * L_ + (e & 31)];   // tile(context, 16)

    // mean
    float sum = y;
    #pragma unroll
    for (int o = 16; o > 0; o >>= 1) sum += __shfl_xor(sum, o, 32);
    if ((e & 31) == 0) red[e >> 5] = sum;
    __syncthreads();
    float mu = 0.f;
    #pragma unroll
    for (int i = 0; i < 16; ++i) mu += red[i];
    mu *= (1.f / E_);
    // variance
    const float dm = y - mu;
    float vs = dm * dm;
    #pragma unroll
    for (int o = 16; o > 0; o >>= 1) vs += __shfl_xor(vs, o, 32);
    __syncthreads();
    if ((e & 31) == 0) red[e >> 5] = vs;
    __syncthreads();
    float var = 0.f;
    #pragma unroll
    for (int i = 0; i < 16; ++i) var += red[i];
    var *= (1.f / E_);

    const float norm = dm * rsqrtf(var + 1e-5f) * lnw[e] + lnb[e];
    fb[(size_t)row * E_ + e] = (__bf16)norm;
}

// ---------------------------------------------------------------------------
// Kernel 5: logits GEMM [4096,512]bf16 x [512,32000] -> f32, + b_out.
// One wave computes a 64x64 tile: 16 K-steps x (4 B loads, 4x(A load + 4 WMMA))
// = 256 WMMAs/wave. 4x less L2 traffic than a 16-row tile.
// ---------------------------------------------------------------------------
__global__ __launch_bounds__(32) void k_logits(
    const __bf16* __restrict__ fb, const __bf16* __restrict__ WoT,
    const float* __restrict__ bout, float* __restrict__ out)
{
    const int lane = threadIdx.x;
    const int half = lane >> 4;
    const int ln   = lane & 15;
    const int n0   = blockIdx.x * 64;          // vocab tile (4 sub-tiles of 16)
    const int m0   = blockIdx.y * 64;          // row tile (4 sub-tiles of 16)

    v8f acc[4][4] = {};                        // [mi][nj], 128 VGPRs
    for (int ks = 0; ks < E_; ks += 32) {
        v16bf Bv[4];
        #pragma unroll
        for (int j = 0; j < 4; ++j)
            Bv[j] = *(const v16bf*)(WoT + (size_t)(n0 + j * 16 + ln) * E_ + ks + half * 16);
        if (ks + 32 < E_)
            __builtin_prefetch(WoT + (size_t)(n0 + ln) * E_ + ks + 32, 0, 1);
        #pragma unroll
        for (int mi = 0; mi < 4; ++mi) {
            const __bf16* arow = fb + (size_t)(m0 + mi * 16 + ln) * E_ + ks;
            ABuild A;
            A.h[0] = *(const v8bf*)(arow + half * 16);
            A.h[1] = *(const v8bf*)(arow + half * 16 + 8);
            #pragma unroll
            for (int j = 0; j < 4; ++j)
                acc[mi][j] = __builtin_amdgcn_wmma_f32_16x16x32_bf16(
                    false, A.v, false, Bv[j], (short)0, acc[mi][j], false, false);
        }
    }

    #pragma unroll
    for (int j = 0; j < 4; ++j) {
        const int v = n0 + j * 16 + ln;
        const float bo = bout[v];
        #pragma unroll
        for (int mi = 0; mi < 4; ++mi) {
            #pragma unroll
            for (int r = 0; r < 8; ++r)
                out[(size_t)(m0 + mi * 16 + r + 8 * half) * V_ + v] = acc[mi][j][r] + bo;
        }
    }
}

// ---------------------------------------------------------------------------
extern "C" void kernel_launch(void* const* d_in, const int* in_sizes, int n_in,
                              void* d_out, int out_size, void* d_ws, size_t ws_size,
                              hipStream_t stream)
{
    const float* Q   = (const float*)d_in[0];
    const float* K   = (const float*)d_in[1];
    const float* Vv  = (const float*)d_in[2];
    const float* X   = (const float*)d_in[3];
    const float* Wd  = (const float*)d_in[4];
    const float* bd  = (const float*)d_in[5];
    const float* Wup = (const float*)d_in[6];
    const float* bup = (const float*)d_in[7];
    const float* lnw = (const float*)d_in[8];
    const float* lnb = (const float*)d_in[9];
    const float* Wo  = (const float*)d_in[10];
    const float* bo  = (const float*)d_in[11];
    float* out = (float*)d_out;

    char* ws = (char*)d_ws;
    const size_t LAT_BF  = (size_t)B_ * S_ * L_ * sizeof(__bf16);   // 256 KB
    const size_t LAT_F32 = (size_t)B_ * S_ * L_ * sizeof(float);    // 512 KB
    __bf16* qb  = (__bf16*)ws;                 ws += LAT_BF;
    __bf16* kb  = (__bf16*)ws;                 ws += LAT_BF;
    __bf16* vT  = (__bf16*)ws;                 ws += LAT_BF;
    float*  xl  = (float*)ws;                  ws += LAT_F32;
    float*  ctx = (float*)ws;                  ws += LAT_F32;
    __bf16* fb  = (__bf16*)ws;                 ws += (size_t)B_ * S_ * E_ * sizeof(__bf16);
    __bf16* WoT = (__bf16*)ws;                 // 32000*512*2 = 32.8 MB

    k_proj  <<<dim3(B_ * S_ / 8), dim3(256),        0, stream>>>(Q, K, Vv, X, Wd, bd, qb, kb, vT, xl);
    k_woutT <<<dim3(V_ / 32, E_ / 32), dim3(32, 8), 0, stream>>>(Wo, WoT);
    k_attn  <<<dim3(S_ / 16, B_), dim3(32),         0, stream>>>(qb, kb, vT, ctx);
    k_upln  <<<dim3(B_ * S_), dim3(512),            0, stream>>>(ctx, xl, Wup, bup, lnw, lnb, fb);
    k_logits<<<dim3(V_ / 64, B_ * S_ / 64), dim3(32), 0, stream>>>(fb, WoT, bo, out);
}
